// ConvDown_39101382263329
// MI455X (gfx1250) — matrix operand order repaired and checked by reading, not compiled, not run on hardware
//
#include <hip/hip_runtime.h>
#include <hip/hip_bf16.h>

typedef __attribute__((ext_vector_type(16))) _Float16 v16h;
typedef __attribute__((ext_vector_type(8)))  _Float16 v8h;
typedef __attribute__((ext_vector_type(4)))  _Float16 v4h;
typedef __attribute__((ext_vector_type(8)))  float    v8f;

#define NPTS      8192
#define NPOINT    2048
#define NSAMPLE   32
#define CIN       64
#define KREAL     67
#define KPAD      96      // 3 x K=32 WMMA steps
#define KSTRIDE   104     // padded LDS row stride (208 B) -> conflict-free
#define OCH       128
#define BATCH     8
#define FPS_BLOCK 1024
#define PPT       8       // points per thread in FPS

// ---------------------------------------------------------------------------
// Kernel 1: Farthest point sampling, one workgroup per batch.
// Points + running distances live in registers; argmax via wave32 shuffles.
// Also emits new_xyz / new_mask / new_key_mask (as f32) directly.
// ---------------------------------------------------------------------------
__global__ __launch_bounds__(FPS_BLOCK) void fps_kernel(
    const float* __restrict__ xyz, const int* __restrict__ mask,
    const int* __restrict__ kmask, float* __restrict__ out_xyz,
    float* __restrict__ out_mask, float* __restrict__ out_kmask)
{
  const int b   = blockIdx.x;
  const int tid = threadIdx.x;
  const int lane = tid & 31;
  const int wv   = tid >> 5;

  const float* X  = xyz   + (size_t)b * NPTS * 3;
  const int*   KM = kmask + (size_t)b * NPTS;
  const int*   MM = mask  + (size_t)b * NPTS;

  __shared__ int   s_hist[NPOINT];
  __shared__ float s_wval[32];
  __shared__ int   s_widx[32];
  __shared__ float s_bc[3];
  __shared__ int   s_next;

  float px[PPT], py[PPT], pz[PPT], dbest[PPT];
#pragma unroll
  for (int i = 0; i < PPT; ++i) {
    int p = i * FPS_BLOCK + tid;
    float sc = (KM[p] != 0) ? ((float)p + 10.0f) * 10.0f : 0.0f;
    px[i] = X[p * 3 + 0] + sc;
    py[i] = X[p * 3 + 1] + sc;
    pz[i] = X[p * 3 + 2] + sc;
    dbest[i] = 1e10f;
  }

  if (tid == 0) { s_bc[0] = px[0]; s_bc[1] = py[0]; s_bc[2] = pz[0]; }
  __syncthreads();

  int last = 0;
  for (int it = 0; it < NPOINT; ++it) {
    if (tid == 0) s_hist[it] = last;
    const float cx = s_bc[0], cy = s_bc[1], cz = s_bc[2];

    float vbest = -1.0f; int ibest = 0;
#pragma unroll
    for (int i = 0; i < PPT; ++i) {
      float dx = px[i] - cx, dy = py[i] - cy, dz = pz[i] - cz;
      float d = dx * dx + dy * dy + dz * dz;
      dbest[i] = fminf(dbest[i], d);
      int p = i * FPS_BLOCK + tid;
      if (dbest[i] > vbest || (dbest[i] == vbest && p < ibest)) {
        vbest = dbest[i]; ibest = p;
      }
    }
    // wave32 argmax (ties -> smaller index, matching jnp.argmax)
#pragma unroll
    for (int off = 16; off > 0; off >>= 1) {
      float ov = __shfl_xor(vbest, off, 32);
      int   oi = __shfl_xor(ibest, off, 32);
      if (ov > vbest || (ov == vbest && oi < ibest)) { vbest = ov; ibest = oi; }
    }
    if (lane == 0) { s_wval[wv] = vbest; s_widx[wv] = ibest; }
    __syncthreads();
    if (wv == 0) {
      vbest = s_wval[lane]; ibest = s_widx[lane];
#pragma unroll
      for (int off = 16; off > 0; off >>= 1) {
        float ov = __shfl_xor(vbest, off, 32);
        int   oi = __shfl_xor(ibest, off, 32);
        if (ov > vbest || (ov == vbest && oi < ibest)) { vbest = ov; ibest = oi; }
      }
      if (lane == 0) s_next = ibest;
    }
    __syncthreads();
    last = s_next;
    if ((last & (FPS_BLOCK - 1)) == tid) {           // owner broadcasts coords
      int i = last >> 10;
      s_bc[0] = px[i]; s_bc[1] = py[i]; s_bc[2] = pz[i];
    }
    __syncthreads();
  }

  for (int t = tid; t < NPOINT; t += FPS_BLOCK) {
    int id = s_hist[t];
    size_t o = (size_t)b * NPOINT + t;
    out_xyz[o * 3 + 0] = X[id * 3 + 0];
    out_xyz[o * 3 + 1] = X[id * 3 + 1];
    out_xyz[o * 3 + 2] = X[id * 3 + 2];
    out_mask[o]  = (float)MM[id];
    out_kmask[o] = (float)KM[id];
  }
}

// ---------------------------------------------------------------------------
// Kernel 2: ball query — wave per query; ballot_w32 + popc prefix gives the
// lowest-index-first 32 neighbors (reference top_k-on-index semantics).
// ---------------------------------------------------------------------------
__global__ __launch_bounds__(256) void ballq_kernel(
    const float* __restrict__ xyz, const float* __restrict__ new_xyz,
    int* __restrict__ ridx)
{
  const int lane = threadIdx.x & 31;
  const int wv   = threadIdx.x >> 5;
  const int qi   = blockIdx.x * 8 + wv;       // 0 .. B*NPOINT-1
  const int b    = qi >> 11;                  // NPOINT = 2048
  const float* X = xyz + (size_t)b * NPTS * 3;

  __shared__ int buf[8][NSAMPLE];

  const float qx = new_xyz[(size_t)qi * 3 + 0];
  const float qy = new_xyz[(size_t)qi * 3 + 1];
  const float qz = new_xyz[(size_t)qi * 3 + 2];
  const float r2 = 0.25f;                     // radius^2

  int cnt = 0;
  for (int base = 0; base < NPTS && cnt < NSAMPLE; base += 32) {
    __builtin_prefetch(X + (size_t)(base + 64) * 3, 0, 1);
    int p = base + lane;
    float dx = X[p * 3 + 0] - qx;
    float dy = X[p * 3 + 1] - qy;
    float dz = X[p * 3 + 2] - qz;
    float d2 = dx * dx + dy * dy + dz * dz;
    bool within = d2 < r2;
    unsigned m = __builtin_amdgcn_ballot_w32(within);
    int pos = cnt + __popc(m & ((1u << lane) - 1u));
    if (within && pos < NSAMPLE) buf[wv][pos] = p;
    cnt += __popc(m);
  }
  int n = cnt < NSAMPLE ? cnt : NSAMPLE;
  int v = (n == 0) ? 0 : ((lane < n) ? buf[wv][lane] : buf[wv][0]);
  ridx[(size_t)qi * NSAMPLE + lane] = v;
}

// ---------------------------------------------------------------------------
// Kernel 3: gather + 1x1 conv (WMMA f32<-f16) + bias + LeakyReLU + mean/max
// pooling fused. One wave per query. W stored K-contiguous per out-channel
// (Wl2[n][k]) and X row-major (Xl[row][k]) so every WMMA fragment is two
// 16-byte contiguous LDS chunks -> ds_load_b128 pairs, with 104-half row
// stride (208 B) making the 16-lane access pattern bank-conflict-free.
// Per query: 2 row-tiles x 8 col-tiles x 3 K-steps = 48 v_wmma ops.
// ---------------------------------------------------------------------------
__global__ __launch_bounds__(128) void conv_kernel(
    const float* __restrict__ feat, const float* __restrict__ xyz,
    const float* __restrict__ new_xyz, const int* __restrict__ ridx,
    const float* __restrict__ W, const float* __restrict__ bias,
    float* __restrict__ out)
{
  __shared__ __align__(16) _Float16 Wl2[OCH][KSTRIDE];        // 26 KB (W^T rows)
  __shared__ __align__(16) _Float16 Xl[4][NSAMPLE][KSTRIDE];  // 26 KB

  const int tid  = threadIdx.x;
  const int lane = tid & 31;
  const int wv   = tid >> 5;
  const int qi   = blockIdx.x * 4 + wv;
  const int b    = qi >> 11;

  // Stage W: Wl2[n][k] = W[n*67+k], zero-padded K in [67,96).
  for (int t = tid; t < OCH * KPAD; t += 128) {
    int n = t / KPAD, k = t % KPAD;
    Wl2[n][k] = (k < KREAL) ? (_Float16)W[n * KREAL + k] : (_Float16)0.0f;
  }

  // Stage this wave's X: lane = sample row; channels 0..63 feat, 64..66 dxyz.
  {
    int id = ridx[(size_t)qi * NSAMPLE + lane];
    const float* F = feat + ((size_t)b * NPTS + id) * CIN;
    _Float16* xr = &Xl[wv][lane][0];
#pragma unroll
    for (int c = 0; c < CIN; c += 4) {
      float4 f4 = *reinterpret_cast<const float4*>(F + c);
      v4h h; h.x = (_Float16)f4.x; h.y = (_Float16)f4.y;
      h.z = (_Float16)f4.z; h.w = (_Float16)f4.w;
      *reinterpret_cast<v4h*>(xr + c) = h;
    }
    float qx = new_xyz[(size_t)qi * 3 + 0];
    float qy = new_xyz[(size_t)qi * 3 + 1];
    float qz = new_xyz[(size_t)qi * 3 + 2];
    const float* P = xyz + ((size_t)b * NPTS + id) * 3;
    v8h tail = {};
    tail.s0 = (_Float16)(P[0] - qx);
    tail.s1 = (_Float16)(P[1] - qy);
    tail.s2 = (_Float16)(P[2] - qz);
    *reinterpret_cast<v8h*>(xr + 64) = tail;                  // 64..71 (3 + pad0)
    v8h z = {};
    *reinterpret_cast<v8h*>(xr + 72) = z;
    *reinterpret_cast<v8h*>(xr + 80) = z;
    *reinterpret_cast<v8h*>(xr + 88) = z;
  }
  __syncthreads();

  const bool hi  = lane >= 16;
  const int  rlo = lane & 15;
  const int  aoff = hi ? 8 : 0;    // A frag K sub-offset per lane half
  const int  boff = hi ? 8 : 0;    // B frag K offset per lane half

  for (int nt = 0; nt < 8; ++nt) {              // tiles 0-3: mean half, 4-7: max half
    const int n0 = nt * 16;
    v8f c0 = {}; v8f c1 = {};
    const _Float16* wrow = &Wl2[n0 + rlo][0];
#pragma unroll
    for (int ks = 0; ks < 3; ++ks) {
      const int kb = ks * 32;
      union { v16h v; v8h h8[2]; } a0, a1, bm;
      const _Float16* x0 = &Xl[wv][rlo][kb + aoff];
      const _Float16* x1 = &Xl[wv][rlo + 16][kb + aoff];
      a0.h8[0] = *reinterpret_cast<const v8h*>(x0);
      a0.h8[1] = *reinterpret_cast<const v8h*>(x0 + 16);
      a1.h8[0] = *reinterpret_cast<const v8h*>(x1);
      a1.h8[1] = *reinterpret_cast<const v8h*>(x1 + 16);
      bm.h8[0] = *reinterpret_cast<const v8h*>(wrow + kb + boff);
      bm.h8[1] = *reinterpret_cast<const v8h*>(wrow + kb + boff + 16);
      c0 = __builtin_amdgcn_wmma_f32_16x16x32_f16(false, a0.v, false, bm.v,
                                                  (short)0, c0, false, false);
      c1 = __builtin_amdgcn_wmma_f32_16x16x32_f16(false, a1.v, false, bm.v,
                                                  (short)0, c1, false, false);
    }

    const float bb = bias[n0 + rlo];
    float acc;
    if (nt < 4) {                               // mean over 32 samples
      float s = 0.0f;
#pragma unroll
      for (int i = 0; i < 8; ++i) {
        float y0 = c0[i] + bb; y0 = (y0 > 0.0f) ? y0 : 0.01f * y0;
        float y1 = c1[i] + bb; y1 = (y1 > 0.0f) ? y1 : 0.01f * y1;
        s += y0 + y1;
      }
      s += __shfl_xor(s, 16, 32);               // combine rows across lane halves
      acc = s * (1.0f / 32.0f);
    } else {                                    // max over 32 samples
      float mx = -3.4e38f;
#pragma unroll
      for (int i = 0; i < 8; ++i) {
        float y0 = c0[i] + bb; y0 = (y0 > 0.0f) ? y0 : 0.01f * y0;
        float y1 = c1[i] + bb; y1 = (y1 > 0.0f) ? y1 : 0.01f * y1;
        mx = fmaxf(mx, fmaxf(y0, y1));
      }
      mx = fmaxf(mx, __shfl_xor(mx, 16, 32));
      acc = mx;
    }
    if (lane < 16) out[(size_t)qi * OCH + n0 + lane] = acc;
  }
}

// ---------------------------------------------------------------------------
extern "C" void kernel_launch(void* const* d_in, const int* in_sizes, int n_in,
                              void* d_out, int out_size, void* d_ws, size_t ws_size,
                              hipStream_t stream) {
  const float* feat  = (const float*)d_in[0];   // [B,N,64]
  const float* xyz   = (const float*)d_in[1];   // [B,N,3]
  const int*   mask  = (const int*)d_in[2];     // [B,N]
  const int*   kmask = (const int*)d_in[3];     // [B,N]
  const float* W     = (const float*)d_in[4];   // [128,67]
  const float* bias  = (const float*)d_in[5];   // [128]

  float* out     = (float*)d_out;                        // [B,M,128]
  float* o_xyz   = out   + (size_t)BATCH * NPOINT * OCH; // [B,M,3]
  float* o_mask  = o_xyz + (size_t)BATCH * NPOINT * 3;   // [B,M]
  float* o_kmask = o_mask + (size_t)BATCH * NPOINT;      // [B,M]

  int* ridx = (int*)d_ws;                                // [B,M,32] = 2 MB

  fps_kernel<<<BATCH, FPS_BLOCK, 0, stream>>>(xyz, mask, kmask,
                                              o_xyz, o_mask, o_kmask);
  ballq_kernel<<<(BATCH * NPOINT) / 8, 256, 0, stream>>>(xyz, o_xyz, ridx);
  conv_kernel<<<(BATCH * NPOINT) / 4, 128, 0, stream>>>(feat, xyz, o_xyz, ridx,
                                                        W, bias, out);
  (void)in_sizes; (void)n_in; (void)out_size; (void)ws_size;
}
